// HybridViTGATBlock_17343077941961
// MI455X (gfx1250) — compile-verified, matrix operand order checked by reading, not run on hardware
//
#include <hip/hip_runtime.h>
#include <hip/hip_bf16.h>
#include <math.h>

// ---------------- WMMA types / fragments (gfx1250, wave32) ----------------
typedef __bf16 bf16x16 __attribute__((ext_vector_type(16)));
typedef float  f32x8   __attribute__((ext_vector_type(8)));

union FragBF {
  bf16x16 v;
  unsigned int   d[8];
  unsigned short u[16];
};

__device__ __forceinline__ unsigned short f2bf(float x) {
  unsigned int u = __float_as_uint(x);
  unsigned int r = u + 0x7FFFu + ((u >> 16) & 1u);   // round-to-nearest-even
  return (unsigned short)(r >> 16);
}

// A (16x32 bf16) / Bt (transposed-to-[N,K]) fragment loader: pairs contiguous.
// ISA 16-bit A layout: lane m = lane&15; K-half = (lane>>4)*8;
// VGPR v: k = (v>=4?16:0) + khalf + (v&3)*2, holds (k, k+1).
__device__ __forceinline__ FragBF load_frag_rowpair(const unsigned short* base,
                                                    int ld, int row0, int col0,
                                                    int lane) {
  FragBF f;
  int m  = lane & 15;
  int kb = (lane >> 4) * 8;
  const unsigned short* p = base + (size_t)(row0 + m) * ld + col0;
#pragma unroll
  for (int v = 0; v < 8; ++v) {
    int k = ((v >= 4) ? 16 : 0) + kb + (v & 3) * 2;
    f.d[v] = *(const unsigned int*)(p + k);
  }
  return f;
}

__device__ __forceinline__ f32x8 wmma_bf16(const FragBF& a, const FragBF& b, f32x8 c) {
  return __builtin_amdgcn_wmma_f32_16x16x32_bf16(false, a.v, false, b.v,
                                                 (short)0, c, false, false);
}

// gfx1250 async global -> LDS DMA (B128), tracked by ASYNCcnt.
// VDST = per-lane LDS byte address VGPR, VADDR = 64-bit global address.
__device__ __forceinline__ void async_g2l_b128(unsigned int lds_off, const void* gaddr) {
  asm volatile("global_load_async_to_lds_b128 %0, %1, off"
               :: "v"(lds_off), "v"(gaddr) : "memory");
}
__device__ __forceinline__ void wait_asynccnt0() {
  asm volatile("s_wait_asynccnt 0" ::: "memory");
}

// ---- Generic bf16 WMMA GEMM, 2x2 register-blocked (wave computes 32x32) ----
// A: [M,K] bf16 row-major.  Bt: [N,K] bf16 row-major (weight transposed).
// C = act(A*Bt^T + bias) + residual ; optional f32 and bf16 outputs.
__global__ __launch_bounds__(32)
void wmma_gemm_kernel(const unsigned short* __restrict__ A,
                      const unsigned short* __restrict__ Bt,
                      const float* __restrict__ bias,
                      const float* __restrict__ residual,
                      float* __restrict__ Cf32,
                      unsigned short* __restrict__ Cbf,
                      int M, int N, int K, int act) {
  int lane  = threadIdx.x;
  int nbase = blockIdx.x * 32;
  int mbase = blockIdx.y * 32;
  f32x8 acc00 = {}, acc01 = {}, acc10 = {}, acc11 = {};
  for (int kk = 0; kk < K; kk += 32) {
    FragBF a0 = load_frag_rowpair(A,  K, mbase,      kk, lane);
    FragBF a1 = load_frag_rowpair(A,  K, mbase + 16, kk, lane);
    FragBF b0 = load_frag_rowpair(Bt, K, nbase,      kk, lane);
    FragBF b1 = load_frag_rowpair(Bt, K, nbase + 16, kk, lane);
    acc00 = wmma_bf16(a0, b0, acc00);
    acc01 = wmma_bf16(a0, b1, acc01);
    acc10 = wmma_bf16(a1, b0, acc10);
    acc11 = wmma_bf16(a1, b1, acc11);
  }
  int rowoff = (lane < 16) ? 0 : 8;
  int coll   = lane & 15;
  auto epi = [&](f32x8 acc, int mo, int no) {
    int col = nbase + no + coll;
    float bv = bias ? bias[col] : 0.0f;
#pragma unroll
    for (int r = 0; r < 8; ++r) {
      int row = mbase + mo + r + rowoff;
      float v = acc[r] + bv;
      if (act == 1) v = 0.5f * v * (1.0f + erff(v * 0.70710678118654752f)); // GELU
      if (residual) v += residual[(size_t)row * N + col];
      if (Cf32) Cf32[(size_t)row * N + col] = v;
      if (Cbf)  Cbf [(size_t)row * N + col] = f2bf(v);
    }
  };
  epi(acc00, 0, 0);  epi(acc01, 0, 16);
  epi(acc10, 16, 0); epi(acc11, 16, 16);
}

// ---------------- Flash attention (one wave = 16 queries of one head) -------
// Q/K/V: [Ns, E] bf16, head h occupies cols [h*32, h*32+32). O: [Ns, E] bf16.
// V tile (32 kv rows x 32 cols) staged to LDS via async global->LDS DMA.
__global__ __launch_bounds__(32)
void flash_attn_kernel(const unsigned short* __restrict__ Q,
                       const unsigned short* __restrict__ Kb,
                       const unsigned short* __restrict__ V,
                       unsigned short* __restrict__ O,
                       int Nq, int E) {
  __shared__ float          s_s[16 * 32];
  __shared__ unsigned short s_p[16 * 32];
  __shared__ unsigned short s_v[32 * 32];
  __shared__ float          s_scale[16];

  int lane  = threadIdx.x;
  int qbase = blockIdx.z * Nq + blockIdx.x * 16;
  int hcol  = blockIdx.y * 32;
  int kv0   = blockIdx.z * Nq;
  const float scale = 0.17677669529663687f; // 1/sqrt(32)

  FragBF qf = load_frag_rowpair(Q, E, qbase, hcol, lane);

  f32x8 o0 = {}, o1 = {};
  float m_run = -3.0e38f, l_run = 0.0f;
  int colw   = lane & 15;
  int rowoff = (lane < 16) ? 0 : 8;

  for (int c = 0; c < Nq; c += 32) {
    // ---- stage V tile [32 x 32] into LDS with async DMA (ASYNCcnt path)
#pragma unroll
    for (int j = 0; j < 4; ++j) {
      int chunk = j * 32 + lane;           // 128 chunks x 16B = 2KB tile
      int rr = chunk >> 2;                 // kv row within tile
      int cc = (chunk & 3) * 8;            // bf16 col offset within tile
      const unsigned short* gp = V + (size_t)(kv0 + c + rr) * E + hcol + cc;
      unsigned int loff = (unsigned int)(unsigned long long)(const void*)&s_v[rr * 32 + cc];
      async_g2l_b128(loff, gp);
    }

    // ---- S = Q * K^T : Bt-style contiguous loader on K rows
    FragBF k0 = load_frag_rowpair(Kb, E, kv0 + c,      hcol, lane);
    FragBF k1 = load_frag_rowpair(Kb, E, kv0 + c + 16, hcol, lane);
    f32x8 s0 = {}, s1 = {};
    s0 = wmma_bf16(qf, k0, s0);
    s1 = wmma_bf16(qf, k1, s1);
#pragma unroll
    for (int r = 0; r < 8; ++r) {
      s_s[(r + rowoff) * 32 + colw]      = s0[r] * scale;
      s_s[(r + rowoff) * 32 + colw + 16] = s1[r] * scale;
    }
    __syncthreads();
    if (lane < 16) {                       // per-row online softmax
      float cmax = -3.0e38f;
#pragma unroll
      for (int j = 0; j < 32; ++j) cmax = fmaxf(cmax, s_s[lane * 32 + j]);
      float mn = fmaxf(m_run, cmax);
      float sc = __expf(m_run - mn);
      float ls = 0.0f;
#pragma unroll
      for (int j = 0; j < 32; ++j) {
        float p = __expf(s_s[lane * 32 + j] - mn);
        ls += p;
        s_p[lane * 32 + j] = f2bf(p);
      }
      l_run = l_run * sc + ls;
      m_run = mn;
      s_scale[lane] = sc;
    }
    __syncthreads();
#pragma unroll
    for (int r = 0; r < 8; ++r) {          // rescale running O
      float sc = s_scale[r + rowoff];
      o0[r] *= sc; o1[r] *= sc;
    }
    FragBF pf;                              // P fragment from LDS (A-layout)
    {
      int m  = lane & 15;
      int kb = (lane >> 4) * 8;
#pragma unroll
      for (int vv = 0; vv < 8; ++vv) {
        int k = ((vv >= 4) ? 16 : 0) + kb + (vv & 3) * 2;
        pf.u[2 * vv]     = s_p[m * 32 + k];
        pf.u[2 * vv + 1] = s_p[m * 32 + k + 1];
      }
    }
    wait_asynccnt0();                       // V tile resident in LDS
    FragBF v0, v1;                          // B fragments read transposed from LDS
    {
      int n  = lane & 15;
      int kb = (lane >> 4) * 8;
#pragma unroll
      for (int vv = 0; vv < 8; ++vv) {
        int k = ((vv >= 4) ? 16 : 0) + kb + (vv & 3) * 2;
        v0.u[2 * vv]     = s_v[k * 32 + n];
        v0.u[2 * vv + 1] = s_v[(k + 1) * 32 + n];
        v1.u[2 * vv]     = s_v[k * 32 + 16 + n];
        v1.u[2 * vv + 1] = s_v[(k + 1) * 32 + 16 + n];
      }
    }
    o0 = wmma_bf16(pf, v0, o0);
    o1 = wmma_bf16(pf, v1, o1);
    __syncthreads();
  }
  if (lane < 16) s_scale[lane] = 1.0f / l_run;
  __syncthreads();
#pragma unroll
  for (int r = 0; r < 8; ++r) {
    int row = qbase + r + rowoff;
    float inv = s_scale[r + rowoff];
    O[(size_t)row * E + hcol + colw]      = f2bf(o0[r] * inv);
    O[(size_t)row * E + hcol + 16 + colw] = f2bf(o1[r] * inv);
  }
}

// ---------------- tokenize [B,C,H,W] -> [Ns,C] + LayerNorm (bf16 out) -------
__global__ __launch_bounds__(256)
void tokenize_ln1_kernel(const float* __restrict__ xin,
                         const float* __restrict__ g, const float* __restrict__ b,
                         unsigned short* __restrict__ out,
                         int C, int HW) {
  __shared__ float red[256];
  int tok = blockIdx.x, t = threadIdx.x;
  int bi = tok / HW, n = tok % HW;
  float x = xin[((size_t)bi * C + t) * HW + n];
  red[t] = x; __syncthreads();
  for (int s = 128; s > 0; s >>= 1) { if (t < s) red[t] += red[t + s]; __syncthreads(); }
  float mu = red[0] * (1.0f / 256.0f); __syncthreads();
  float d = x - mu;
  red[t] = d * d; __syncthreads();
  for (int s = 128; s > 0; s >>= 1) { if (t < s) red[t] += red[t + s]; __syncthreads(); }
  float var = red[0] * (1.0f / 256.0f);
  out[(size_t)tok * 256 + t] = f2bf(d * rsqrtf(var + 1e-5f) * g[t] + b[t]);
}

// ---------------- generic row LayerNorm (256 cols) --------------------------
__global__ __launch_bounds__(256)
void ln_rows_kernel(const float* __restrict__ in,
                    const float* __restrict__ g, const float* __restrict__ b,
                    float* __restrict__ of32, unsigned short* __restrict__ obf) {
  __shared__ float red[256];
  int row = blockIdx.x, t = threadIdx.x;
  float x = in[(size_t)row * 256 + t];
  red[t] = x; __syncthreads();
  for (int s = 128; s > 0; s >>= 1) { if (t < s) red[t] += red[t + s]; __syncthreads(); }
  float mu = red[0] * (1.0f / 256.0f); __syncthreads();
  float d = x - mu;
  red[t] = d * d; __syncthreads();
  for (int s = 128; s > 0; s >>= 1) { if (t < s) red[t] += red[t + s]; __syncthreads(); }
  float var = red[0] * (1.0f / 256.0f);
  float y = d * rsqrtf(var + 1e-5f) * g[t] + b[t];
  if (of32) of32[(size_t)row * 256 + t] = y;
  if (obf)  obf [(size_t)row * 256 + t] = f2bf(y);
}

// ---------------- final LayerNorm + NCHW store ------------------------------
__global__ __launch_bounds__(256)
void final_ln_kernel(const float* __restrict__ in,
                     const float* __restrict__ g, const float* __restrict__ b,
                     float* __restrict__ out, int E, int HW) {
  __shared__ float red[256];
  int row = blockIdx.x, t = threadIdx.x;
  int bi = row / HW, n = row % HW;
  float x = in[(size_t)row * 256 + t];
  red[t] = x; __syncthreads();
  for (int s = 128; s > 0; s >>= 1) { if (t < s) red[t] += red[t + s]; __syncthreads(); }
  float mu = red[0] * (1.0f / 256.0f); __syncthreads();
  float d = x - mu;
  red[t] = d * d; __syncthreads();
  for (int s = 128; s > 0; s >>= 1) { if (t < s) red[t] += red[t + s]; __syncthreads(); }
  float var = red[0] * (1.0f / 256.0f);
  out[((size_t)bi * E + t) * HW + n] = d * rsqrtf(var + 1e-5f) * g[t] + b[t];
}

// ---------------- GAT attention scalars: a_src/a_dst per node ---------------
__global__ __launch_bounds__(256)
void gat_scores_kernel(const float* __restrict__ h,
                       const float* __restrict__ as, const float* __restrict__ ad,
                       float* __restrict__ osrc, float* __restrict__ odst) {
  __shared__ float red[256];
  int row = blockIdx.x, t = threadIdx.x;
  float hv = h[(size_t)row * 256 + t];
  red[t] = hv * as[t]; __syncthreads();
  for (int s = 128; s > 0; s >>= 1) { if (t < s) red[t] += red[t + s]; __syncthreads(); }
  if (t == 0) osrc[row] = red[0];
  __syncthreads();
  red[t] = hv * ad[t]; __syncthreads();
  for (int s = 128; s > 0; s >>= 1) { if (t < s) red[t] += red[t + s]; __syncthreads(); }
  if (t == 0) odst[row] = red[0];
}

__device__ __forceinline__ void atomicMaxF(float* addr, float val) {
  int* ia = (int*)addr;
  int old = __float_as_int(*addr);
  while (__int_as_float(old) < val) {
    int assumed = old;
    old = atomicCAS(ia, assumed, __float_as_int(val));
    if (old == assumed) break;
  }
}

__global__ void edge_logit_max_kernel(const int* __restrict__ src, const int* __restrict__ dst,
                                      const float* __restrict__ as, const float* __restrict__ ad,
                                      float* __restrict__ segm, int Eed) {
  int e = blockIdx.x * blockDim.x + threadIdx.x;
  if (e >= Eed) return;
  float l = as[src[e]] + ad[dst[e]];
  l = (l > 0.0f) ? l : 0.2f * l;             // leaky_relu 0.2
  atomicMaxF(&segm[dst[e]], l);
}

__global__ void edge_exp_sum_kernel(const int* __restrict__ src, const int* __restrict__ dst,
                                    const float* __restrict__ as, const float* __restrict__ ad,
                                    const float* __restrict__ segm,
                                    float* __restrict__ eex, float* __restrict__ segden, int Eed) {
  int e = blockIdx.x * blockDim.x + threadIdx.x;
  if (e >= Eed) return;
  float l = as[src[e]] + ad[dst[e]];
  l = (l > 0.0f) ? l : 0.2f * l;
  float ex = __expf(l - segm[dst[e]]);
  eex[e] = ex;
  atomicAdd(&segden[dst[e]], ex);
}

__global__ __launch_bounds__(256)
void edge_agg_kernel(const int* __restrict__ src, const int* __restrict__ dst,
                     const float* __restrict__ eex, const float* __restrict__ segden,
                     const float* __restrict__ h, float* __restrict__ agg) {
  int e = blockIdx.x, c = threadIdx.x;
  int s = src[e], d = dst[e];
  float coef = eex[e] / (segden[d] + 1e-16f);
  atomicAdd(&agg[(size_t)d * 256 + c], h[(size_t)s * 256 + c] * coef);
}

__global__ __launch_bounds__(256)
void gat_epi_kernel(const float* __restrict__ agg, const float* __restrict__ gbias,
                    const float* __restrict__ nf, float* __restrict__ xag) {
  size_t i = (size_t)blockIdx.x * 256 + threadIdx.x;
  float v = agg[i] + gbias[i & 255];
  v = (v > 0.0f) ? v : expm1f(v);            // ELU
  xag[i] = nf[i] + v;
}

__global__ void fill_f32_kernel(float* __restrict__ p, float v, int n) {
  int i = blockIdx.x * blockDim.x + threadIdx.x;
  if (i < n) p[i] = v;
}

// Transpose + convert 256x256 f32 weight [K,N] -> bf16 [N,K]
__global__ __launch_bounds__(256)
void convert_wT_kernel(const float* __restrict__ w, unsigned short* __restrict__ wt) {
  int idx = blockIdx.x * 256 + threadIdx.x;  // grid 256 -> 65536 elements
  int k = idx >> 8, n = idx & 255;
  wt[n * 256 + k] = f2bf(w[idx]);
}

// ---------------------------------------------------------------------------
extern "C" void kernel_launch(void* const* d_in, const int* in_sizes, int n_in,
                              void* d_out, int out_size, void* d_ws, size_t ws_size,
                              hipStream_t stream) {
  const int Bsz = 2, Hh = 48, Ww = 48, Cc = 256, Ee = 256;
  const int HW = Hh * Ww;          // 2304
  const int Ns = Bsz * HW;         // 4608
  const size_t NsE = (size_t)Ns * Ee;
  const int Eed = in_sizes[1] / 2;

  const float* x_feat = (const float*)d_in[0];
  const int*   eidx   = (const int*)d_in[1];
  const float* n1g = (const float*)d_in[2];
  const float* n1b = (const float*)d_in[3];
  const float* proj_w = (const float*)d_in[4];
  const float* proj_b = (const float*)d_in[5];
  const float* wq = (const float*)d_in[6];  const float* bq = (const float*)d_in[7];
  const float* wk = (const float*)d_in[8];  const float* bk = (const float*)d_in[9];
  const float* wv = (const float*)d_in[10]; const float* bv = (const float*)d_in[11];
  const float* wo = (const float*)d_in[12]; const float* bo = (const float*)d_in[13];
  const float* n2g = (const float*)d_in[14]; const float* n2b = (const float*)d_in[15];
  const float* gpl_w = (const float*)d_in[16]; const float* gpl_b = (const float*)d_in[17];
  const float* gat_w = (const float*)d_in[18];
  const float* att_src = (const float*)d_in[19];
  const float* att_dst = (const float*)d_in[20];
  const float* gat_bias = (const float*)d_in[21];
  const float* n3g = (const float*)d_in[22]; const float* n3b = (const float*)d_in[23];
  const float* mw1 = (const float*)d_in[24]; const float* mb1 = (const float*)d_in[25];
  const float* mw2 = (const float*)d_in[26]; const float* mb2 = (const float*)d_in[27];
  const float* fng = (const float*)d_in[28]; const float* fnb = (const float*)d_in[29];

  // -------- workspace carve --------
  char* wsp = (char*)d_ws;
  auto carve = [&](size_t bytes) -> char* {
    char* p = wsp;
    wsp += (bytes + 255) & ~(size_t)255;
    return p;
  };
  unsigned short* wT[9];
  for (int i = 0; i < 9; ++i) wT[i] = (unsigned short*)carve(65536 * 2);
  unsigned short* xln1 = (unsigned short*)carve(NsE * 2);
  float*          xv   = (float*)carve(NsE * 4);
  unsigned short* xvb  = (unsigned short*)carve(NsE * 2);
  unsigned short* qb   = (unsigned short*)carve(NsE * 2);
  unsigned short* kb   = (unsigned short*)carve(NsE * 2);
  unsigned short* vb   = (unsigned short*)carve(NsE * 2);
  unsigned short* ob   = (unsigned short*)carve(NsE * 2);
  float*          nf   = (float*)carve(NsE * 4);
  unsigned short* ln2b = (unsigned short*)carve(NsE * 2);
  unsigned short* xgb  = (unsigned short*)carve(NsE * 2);
  float*          hft  = (float*)carve(NsE * 4);
  float* asrc = (float*)carve(Ns * 4);
  float* adst = (float*)carve(Ns * 4);
  float* segm = (float*)carve(Ns * 4);
  float* segd = (float*)carve(Ns * 4);
  float* eex  = (float*)carve((size_t)Eed * 4);
  float*          agg  = (float*)carve(NsE * 4);
  float*          xag  = (float*)carve(NsE * 4);
  float*          xn   = (float*)carve(NsE * 4);
  unsigned short* xnb  = (unsigned short*)carve(NsE * 2);
  unsigned short* m1b  = (unsigned short*)carve(NsE * 2);
  float*          prel = (float*)carve(NsE * 4);

  // -------- weights -> bf16 transposed --------
  const float* wsrc[9] = {proj_w, wq, wk, wv, wo, gpl_w, gat_w, mw1, mw2};
  for (int i = 0; i < 9; ++i)
    convert_wT_kernel<<<256, 256, 0, stream>>>(wsrc[i], wT[i]);

  // -------- tokenize + LN1 --------
  tokenize_ln1_kernel<<<Ns, 256, 0, stream>>>(x_feat, n1g, n1b, xln1, Cc, HW);

  dim3 gg(Ee / 32, Ns / 32);   // (8, 144), 32x32 tile per wave
  // proj: xv = LN1(x) @ proj_w + proj_b   (f32 + bf16)
  wmma_gemm_kernel<<<gg, 32, 0, stream>>>(xln1, wT[0], proj_b, nullptr, xv, xvb,
                                          Ns, Ee, Cc, 0);
  // q/k/v
  wmma_gemm_kernel<<<gg, 32, 0, stream>>>(xvb, wT[1], bq, nullptr, nullptr, qb, Ns, Ee, Ee, 0);
  wmma_gemm_kernel<<<gg, 32, 0, stream>>>(xvb, wT[2], bk, nullptr, nullptr, kb, Ns, Ee, Ee, 0);
  wmma_gemm_kernel<<<gg, 32, 0, stream>>>(xvb, wT[3], bv, nullptr, nullptr, vb, Ns, Ee, Ee, 0);

  // flash attention: grid (qtiles, heads, batch)
  flash_attn_kernel<<<dim3(HW / 16, 8, Bsz), 32, 0, stream>>>(qb, kb, vb, ob, HW, Ee);

  // nf = xv + o @ wo + bo
  wmma_gemm_kernel<<<gg, 32, 0, stream>>>(ob, wT[4], bo, xv, nf, nullptr, Ns, Ee, Ee, 0);

  // LN2 -> bf16, gpl, gat
  ln_rows_kernel<<<Ns, 256, 0, stream>>>(nf, n2g, n2b, nullptr, ln2b);
  wmma_gemm_kernel<<<gg, 32, 0, stream>>>(ln2b, wT[5], gpl_b, nullptr, nullptr, xgb, Ns, Ee, Ee, 0);
  wmma_gemm_kernel<<<gg, 32, 0, stream>>>(xgb, wT[6], nullptr, nullptr, hft, nullptr, Ns, Ee, Ee, 0);

  // GAT scalar path
  gat_scores_kernel<<<Ns, 256, 0, stream>>>(hft, att_src, att_dst, asrc, adst);
  fill_f32_kernel<<<(Ns + 255) / 256, 256, 0, stream>>>(segm, -3.0e38f, Ns);
  fill_f32_kernel<<<(Ns + 255) / 256, 256, 0, stream>>>(segd, 0.0f, Ns);
  fill_f32_kernel<<<(int)((NsE + 255) / 256), 256, 0, stream>>>(agg, 0.0f, (int)NsE);
  const int* esrc = eidx;
  const int* edst = eidx + Eed;
  edge_logit_max_kernel<<<(Eed + 255) / 256, 256, 0, stream>>>(esrc, edst, asrc, adst, segm, Eed);
  edge_exp_sum_kernel<<<(Eed + 255) / 256, 256, 0, stream>>>(esrc, edst, asrc, adst, segm, eex, segd, Eed);
  edge_agg_kernel<<<Eed, 256, 0, stream>>>(esrc, edst, eex, segd, hft, agg);
  gat_epi_kernel<<<(int)(NsE / 256), 256, 0, stream>>>(agg, gat_bias, nf, xag);

  // LN3 -> (f32, bf16); MLP with GELU; residual; final LN + NCHW store
  ln_rows_kernel<<<Ns, 256, 0, stream>>>(xag, n3g, n3b, xn, xnb);
  wmma_gemm_kernel<<<gg, 32, 0, stream>>>(xnb, wT[7], mb1, nullptr, nullptr, m1b, Ns, Ee, Ee, 1);
  wmma_gemm_kernel<<<gg, 32, 0, stream>>>(m1b, wT[8], mb2, xn, prel, nullptr, Ns, Ee, Ee, 0);
  final_ln_kernel<<<Ns, 256, 0, stream>>>(prel, fng, fnb, (float*)d_out, Ee, HW);
}
